// GRU_87969520156811
// MI455X (gfx1250) — matrix operand order repaired and checked by reading, not compiled, not run on hardware
//
#include <hip/hip_runtime.h>
#include <cstdint>
#include <cstddef>

#define DEVFN __device__ __forceinline__

typedef __bf16 bf16;
typedef __attribute__((ext_vector_type(16))) __bf16 v16bf;
typedef __attribute__((ext_vector_type(8)))  __bf16 v8bf;
typedef __attribute__((ext_vector_type(8)))  float  v8f;

static constexpr int B  = 64;
static constexpr int T  = 512;
static constexpr int I  = 512;
static constexpr int H  = 1024;
static constexpr int BT = B * T;

union Frag16 { v16bf v; v8bf h[2]; };

// Native cast -> hardware bf16 convert (RNE) instead of 4-op bit twiddle.
DEVFN bf16 f32_to_bf16(float f) { return (bf16)f; }

DEVFN float sigmoidf(float x) { return 1.0f / (1.0f + __expf(-x)); }

// B operand (32x16 bf16): lane n (0..15) holds W[n][k0..k0+7] & W[n][k0+16..k0+23],
// lanes 16..31 hold W[n][k0+8..k0+15] & W[n][k0+24..k0+31]. Caller passes pre-offset pointer.
DEVFN v16bf load_b16(const bf16* p) {
  Frag16 b;
  b.h[0] = ((const v8bf*)p)[0];
  b.h[1] = ((const v8bf*)p)[2];
  return b.v;
}

// A operand (16x32 bf16) direct from a bf16 row in memory/LDS:
// lane<16 -> K 0..7 & 16..23, lane>=16 -> K 8..15 & 24..31 (koff = 0 or 8).
DEVFN v16bf load_a_bf16(const bf16* rowPtr, int koff) {
  Frag16 a;
  a.h[0] = *(const v8bf*)(rowPtr + koff);
  a.h[1] = *(const v8bf*)(rowPtr + koff + 16);
  return a.v;
}

// A operand from an fp32 row (LDS), converting to bf16 on the fly.
DEVFN v16bf cvt_a_f32(const float* rowPtr, int koff) {
  Frag16 a;
#pragma unroll
  for (int j = 0; j < 8; ++j) a.v[j]     = f32_to_bf16(rowPtr[koff + j]);
#pragma unroll
  for (int j = 0; j < 8; ++j) a.v[8 + j] = f32_to_bf16(rowPtr[koff + 16 + j]);
  return a.v;
}

DEVFN v8f wmma_bf16(v16bf a, v16bf b, v8f c) {
  return __builtin_amdgcn_wmma_f32_16x16x32_bf16(false, a, false, b, (short)0, c,
                                                 false, false);
}

// CDNA5 async copy: one 16B chunk per lane, global -> LDS, tracked by ASYNCcnt.
DEVFN void async_b128_to_lds(const float* gsrc, float* lds_dst) {
  uint32_t l = (uint32_t)(uintptr_t)lds_dst;     // low 32 bits == LDS address
  asm volatile("global_load_async_to_lds_b128 %0, %1, off"
               :: "v"(l), "v"(gsrc) : "memory");
}

DEVFN void wait_asynccnt0() {
  asm volatile("s_wait_asynccnt 0x0" ::: "memory");
}

// ---------------------------------------------------------------- converts
__global__ __launch_bounds__(256) void convert_kernel(const float* __restrict__ src,
                                                      bf16* __restrict__ dst, int n) {
  int i = blockIdx.x * blockDim.x + threadIdx.x;
  if (i < n) dst[i] = f32_to_bf16(src[i]);
}

__global__ __launch_bounds__(256) void init_state_kernel(const float* __restrict__ s0,
                                                         bf16* __restrict__ sb16,
                                                         unsigned* __restrict__ ctr, int n) {
  int i = blockIdx.x * blockDim.x + threadIdx.x;
  if (i == 0) *ctr = 0u;                         // reset grid barrier every call
  if (i < n) sb16[i] = f32_to_bf16(s0[i]);
}

// ------------------------------------------------- input projection GEMM
// grid = (BT/64, H/32), block = 256 (8 waves). Wave (mi,ni) owns a 16x16 tile
// of all three gates. The fp32 x tile is streamed into a double-buffered LDS
// staging area with global_load_async_to_lds_b128 (ASYNCcnt), overlapping the
// next tile's fetch with the current tile's WMMAs; fp32->bf16 conversion
// happens at A-fragment read time. Row stride 36 floats keeps 16B alignment
// and conflict-free banks.
__global__ __launch_bounds__(256) void proj_kernel(
    const float* __restrict__ x,
    const bf16* __restrict__ Wzb, const bf16* __restrict__ Wrb, const bf16* __restrict__ Wsb,
    const float* __restrict__ bz, const float* __restrict__ br, const float* __restrict__ bs,
    float* __restrict__ xz, float* __restrict__ xr, float* __restrict__ xs)
{
  __shared__ float Xs[2][64][36];
  const int lane = threadIdx.x & 31;
  const int wave = threadIdx.x >> 5;
  const int mi   = wave >> 1, ni = wave & 1;
  const int bt0  = blockIdx.x * 64;
  const int ncol = blockIdx.y * 32 + ni * 16 + (lane & 15);
  const int rowa = mi * 16 + (lane & 15);
  const int koff = (lane >> 4) * 8;
  const int bsel = (lane >> 4) * 8;
  const int ldr  = threadIdx.x >> 2;         // 0..63
  const int ldc  = (threadIdx.x & 3) * 8;    // 0,8,16,24

  // prologue: stage tile kk=0 into buffer 0
  {
    const float* g = x + (size_t)(bt0 + ldr) * I + ldc;
    float* d = &Xs[0][ldr][ldc];
    async_b128_to_lds(g, d);
    async_b128_to_lds(g + 4, d + 4);
  }
  wait_asynccnt0();
  __syncthreads();

  v8f az = {}, ar = {}, ag = {};

#pragma unroll 2
  for (int kk = 0; kk < I; kk += 32) {
    const int cur = (kk >> 5) & 1;
    if (kk + 32 < I) {                       // stream next tile into other buffer
      const float* g = x + (size_t)(bt0 + ldr) * I + (kk + 32) + ldc;
      float* d = &Xs[cur ^ 1][ldr][ldc];
      async_b128_to_lds(g, d);
      async_b128_to_lds(g + 4, d + 4);
    }
    v16bf a  = cvt_a_f32(&Xs[cur][rowa][0], koff);
    v16bf wz = load_b16(Wzb + (size_t)ncol * I + kk + bsel);
    v16bf wr = load_b16(Wrb + (size_t)ncol * I + kk + bsel);
    v16bf ws = load_b16(Wsb + (size_t)ncol * I + kk + bsel);
    az = wmma_bf16(a, wz, az);
    ar = wmma_bf16(a, wr, ar);
    ag = wmma_bf16(a, ws, ag);
    wait_asynccnt0();
    __syncthreads();
  }

  const int hi8 = (lane >> 4) * 8;
  const float bzv = bz[ncol], brv = br[ncol], bsv = bs[ncol];
#pragma unroll
  for (int v = 0; v < 8; ++v) {
    const int bt = bt0 + mi * 16 + v + hi8;
    const size_t o = (size_t)bt * H + ncol;
    xz[o] = az[v] + bzv;
    xr[o] = ar[v] + brv;
    xs[o] = ag[v] + bsv;
  }
}

// --------------------------------------------------------- grid barrier
DEVFN void grid_barrier(unsigned* ctr, unsigned target) {
  __syncthreads();
  if (threadIdx.x == 0) {
    __hip_atomic_fetch_add(ctr, 1u, __ATOMIC_RELEASE, __HIP_MEMORY_SCOPE_AGENT);
    while (__hip_atomic_load(ctr, __ATOMIC_RELAXED, __HIP_MEMORY_SCOPE_AGENT) < target) {}
    (void)__hip_atomic_fetch_add(ctr, 0u, __ATOMIC_ACQUIRE, __HIP_MEMORY_SCOPE_AGENT);
  }
  __syncthreads();
}

// ------------------------------------------------- persistent recurrence
// grid = 32 blocks (H/32), block = 256. Loops over T internally with a
// device-scope barrier between the (z,r) phase and the s_hat/update phase.
// Per-(m,n) private values (fp32 state s, gate z) stay in registers across
// the whole scan -- only the cross-thread bf16 state (sb16/srb) and the U
// matrices touch memory. A fragments load directly from L2-resident buffers:
// no LDS, no intra-block barriers, loads pipeline across unrolled K iters.
// Epilogue operands (xz/xr/xs/mask) are prefetched before each GEMM loop so
// their latency hides under the 32 K-iterations.
__global__ __launch_bounds__(256) void recur_kernel(
    const bf16* __restrict__ Uzb, const bf16* __restrict__ Urb, const bf16* __restrict__ Usb,
    const float* __restrict__ xz, const float* __restrict__ xr, const float* __restrict__ xs,
    const float* __restrict__ mask, const float* __restrict__ s0,
    bf16* __restrict__ sb16, bf16* __restrict__ srb,
    float* __restrict__ out, unsigned* __restrict__ ctr)
{
  const int lane = threadIdx.x & 31;
  const int wave = threadIdx.x >> 5;
  const int mi   = wave >> 1, ni = wave & 1;
  const int ncol = blockIdx.x * 32 + ni * 16 + (lane & 15);
  const int rowm = mi * 16 + (lane & 15);        // batch row for the A operand
  const int koff = (lane >> 4) * 8;
  const int bsel = (lane >> 4) * 8;
  const int hi8  = (lane >> 4) * 8;
  const unsigned nb = gridDim.x;
  unsigned bar = 0;

  const bf16* aRowS  = sb16 + (size_t)rowm * H;
  const bf16* aRowSR = srb  + (size_t)rowm * H;
  const bf16* uzRow  = Uzb + (size_t)ncol * H;
  const bf16* urRow  = Urb + (size_t)ncol * H;
  const bf16* usRow  = Usb + (size_t)ncol * H;

  // register-resident fp32 state: rows m = mi*16 + hi8 + v, column ncol
  float sreg[8];
#pragma unroll
  for (int v = 0; v < 8; ++v)
    sreg[v] = s0[(mi * 16 + v + hi8) * H + ncol];

  for (int t = 0; t < T; ++t) {
    // ---- phase 1: z = sig(xz + s@Uz^T), r = sig(xr + s@Ur^T), stage s*r (bf16)
    float xzv[8], xrv[8];
#pragma unroll
    for (int v = 0; v < 8; ++v) {              // prefetch epilogue operands
      const size_t go = ((size_t)(mi * 16 + v + hi8) * T + t) * H + ncol;
      xzv[v] = xz[go];
      xrv[v] = xr[go];
    }
    v8f accz = {}, accr = {};
#pragma unroll 4
    for (int kk = 0; kk < H; kk += 32) {
      v16bf a  = load_a_bf16(aRowS + kk, koff);
      v16bf uz = load_b16(uzRow + kk + bsel);
      v16bf ur = load_b16(urRow + kk + bsel);
      accz = wmma_bf16(a, uz, accz);
      accr = wmma_bf16(a, ur, accr);
    }
    float zreg[8];
#pragma unroll
    for (int v = 0; v < 8; ++v) {
      const int m = mi * 16 + v + hi8;
      zreg[v] = sigmoidf(xzv[v] + accz[v]);
      float r = sigmoidf(xrv[v] + accr[v]);
      srb[m * H + ncol] = f32_to_bf16(sreg[v] * r);
    }
    ++bar; grid_barrier(ctr, nb * bar);

    // ---- phase 2: s_hat = relu(xs + (s*r)@Us^T), blend, mask, emit
    float xsv[8], mv[8];
#pragma unroll
    for (int v = 0; v < 8; ++v) {
      const int m = mi * 16 + v + hi8;
      xsv[v] = xs[((size_t)m * T + t) * H + ncol];
      mv[v]  = mask[m * T + t];
    }
    v8f accs = {};
#pragma unroll 4
    for (int kk = 0; kk < H; kk += 32) {
      v16bf a  = load_a_bf16(aRowSR + kk, koff);
      v16bf us = load_b16(usRow + kk + bsel);
      accs = wmma_bf16(a, us, accs);
    }
#pragma unroll
    for (int v = 0; v < 8; ++v) {
      const int m = mi * 16 + v + hi8;
      const size_t go = ((size_t)m * T + t) * H + ncol;
      float sh = fmaxf(xsv[v] + accs[v], 0.0f);
      float sn = (1.0f - zreg[v]) * sreg[v] + zreg[v] * sh;
      sn = sreg[v] * (1.0f - mv[v]) + sn * mv[v];
      out[go] = sn;
      sb16[m * H + ncol] = f32_to_bf16(sn);
      sreg[v] = sn;
    }
    ++bar; grid_barrier(ctr, nb * bar);
  }
}

// ------------------------------------------------------------------ host
extern "C" void kernel_launch(void* const* d_in, const int* in_sizes, int n_in,
                              void* d_out, int out_size, void* d_ws, size_t ws_size,
                              hipStream_t stream) {
  (void)in_sizes; (void)n_in; (void)out_size; (void)ws_size;
  const float* x    = (const float*)d_in[0];
  const float* mask = (const float*)d_in[1];
  const float* s0   = (const float*)d_in[2];
  const float* Ws   = (const float*)d_in[3];
  const float* Wr   = (const float*)d_in[4];
  const float* Wz   = (const float*)d_in[5];
  const float* Us   = (const float*)d_in[6];
  const float* Ur   = (const float*)d_in[7];
  const float* Uz   = (const float*)d_in[8];
  const float* bs   = (const float*)d_in[9];
  const float* br   = (const float*)d_in[10];
  const float* bz   = (const float*)d_in[11];
  float* out = (float*)d_out;

  char* ws = (char*)d_ws;
  // workspace layout (bytes, all 256-aligned)
  float* xzb = (float*)(ws + 0);                         // [BT,H] fp32
  float* xrb = (float*)(ws + 134217728);
  float* xsb = (float*)(ws + 268435456);
  bf16*  Wzb = (bf16*)(ws + 402653184);                  // [H,I] bf16
  bf16*  Wrb = (bf16*)(ws + 403701760);
  bf16*  Wsb = (bf16*)(ws + 404750336);
  bf16*  Uzb = (bf16*)(ws + 405798912);                  // [H,H] bf16
  bf16*  Urb = (bf16*)(ws + 407896064);
  bf16*  Usb = (bf16*)(ws + 409993216);
  bf16*  sb  = (bf16*)(ws + 412614656);                  // [B,H] bf16 state
  bf16*  srb = (bf16*)(ws + 412745728);                  // [B,H] bf16 s*r
  unsigned* ctr = (unsigned*)(ws + 412876800);

  const int nWI = H * I;    // 524288
  const int nUU = H * H;    // 1048576
  convert_kernel<<<(nWI + 255) / 256, 256, 0, stream>>>(Wz, Wzb, nWI);
  convert_kernel<<<(nWI + 255) / 256, 256, 0, stream>>>(Wr, Wrb, nWI);
  convert_kernel<<<(nWI + 255) / 256, 256, 0, stream>>>(Ws, Wsb, nWI);
  convert_kernel<<<(nUU + 255) / 256, 256, 0, stream>>>(Uz, Uzb, nUU);
  convert_kernel<<<(nUU + 255) / 256, 256, 0, stream>>>(Ur, Urb, nUU);
  convert_kernel<<<(nUU + 255) / 256, 256, 0, stream>>>(Us, Usb, nUU);
  init_state_kernel<<<(B * H + 255) / 256, 256, 0, stream>>>(s0, sb, ctr, B * H);

  proj_kernel<<<dim3(BT / 64, H / 32), 256, 0, stream>>>(
      x, Wzb, Wrb, Wsb, bz, br, bs, xzb, xrb, xsb);

  recur_kernel<<<H / 32, 256, 0, stream>>>(
      Uzb, Urb, Usb, xzb, xrb, xsb, mask, s0, sb, srb, out, ctr);
}
